// CRF_84275848282162
// MI455X (gfx1250) — matrix operand order repaired and checked by reading, not compiled, not run on hardware
//
#include <hip/hip_runtime.h>
#include <hip/hip_fp16.h>

// CRF loss: -(score - partition).mean()
// B=128 batches, S=2048 steps, T=64 tags.

#define CRF_B 128
#define CRF_S 2048
#define CRF_T 64

#define INV_LN2 1.44269504088896340736f
#define LN2     0.69314718055994530942f

typedef __attribute__((ext_vector_type(16))) _Float16 v16h;
typedef __attribute__((ext_vector_type(8)))  float    v8f;

// single-instruction hardware transcendentals (base-2)
__device__ __forceinline__ float fast_log2(float x) {
  return __builtin_amdgcn_logf(x);    // v_log_f32
}
__device__ __forceinline__ float fast_exp2(float x) {
  return __builtin_amdgcn_exp2f(x);   // v_exp_f32
}

// ---------------------------------------------------------------------------
// Kernel 1: gold-path score. One block per batch, stride over S, LDS reduce.
// ---------------------------------------------------------------------------
__global__ __launch_bounds__(256) void crf_score_kernel(
    const float* __restrict__ em, const int* __restrict__ tags,
    const unsigned char* __restrict__ mask, const float* __restrict__ trans,
    const float* __restrict__ start_t, const float* __restrict__ end_t,
    float* __restrict__ score) {
  const int b = blockIdx.x;
  const int tid = threadIdx.x;
  __shared__ float redf[256];
  __shared__ int   redi[256];
  float acc = 0.f;
  int   cnt = 0;
  for (int s = tid; s < CRF_S; s += 256) {
    int mk = mask[(size_t)b * CRF_S + s] ? 1 : 0;
    cnt += mk;
    if (s >= 1 && mk) {
      int tg = tags[(size_t)b * CRF_S + s];
      int tp = tags[(size_t)b * CRF_S + s - 1];
      acc += trans[tg * CRF_T + tp] + em[((size_t)b * CRF_S + s) * CRF_T + tg];
    }
  }
  redf[tid] = acc;
  redi[tid] = cnt;
  __syncthreads();
  for (int off = 128; off > 0; off >>= 1) {
    if (tid < off) { redf[tid] += redf[tid + off]; redi[tid] += redi[tid + off]; }
    __syncthreads();
  }
  if (tid == 0) {
    int t0 = tags[(size_t)b * CRF_S];
    float sc = redf[0] + start_t[t0] + em[(size_t)b * CRF_S * CRF_T + t0];
    int last = redi[0] - 1;
    int tl = tags[(size_t)b * CRF_S + last];
    score[b] = sc + end_t[tl];
  }
}

// ---------------------------------------------------------------------------
// Kernel 2: partition. Base-2 forward algorithm on WMMA, lagged normalization.
//   p[m][i] = 2^(a2[m][i] - rmu[m])  (f16, rmu = one-step-stale row max; exact
//   because the same rmu is added back: a2' = rmu + log2(P*E) + em/ln2)
// Single barrier per step; p_sh / pmax_sh double-buffered to avoid races.
// Emissions stream HBM->LDS via global_load_async_to_lds_b128 (9-slot ring,
// depth 8 steps); ASYNCcnt is in-order so <=14 outstanding => slot resident.
// ---------------------------------------------------------------------------
__global__ __launch_bounds__(128) void crf_partition_kernel(
    const float* __restrict__ em, const unsigned char* __restrict__ mask,
    const float* __restrict__ trans, const float* __restrict__ start_t,
    const float* __restrict__ end_t, float* __restrict__ partition) {
  __shared__ float  em_ring[9][16][68];   // 9 slots x (16 rows x 64 floats, pad)
  __shared__ __half p_sh[2][16][72];      // double-buffered [row m][state k]
  __shared__ float  pmax_sh[2][16][4];    // double-buffered per-wave partials
  __shared__ float  fin_sh[16][68];       // final alpha+end for LSE

  const int tid  = threadIdx.x;
  const int w    = tid >> 5;   // wave / n-tile id 0..3
  const int l    = tid & 31;   // lane
  const int half = l >> 4;     // 0/1
  const int l15  = l & 15;
  const int n    = w * 16 + l15;    // output state j (D-layout column)
  const int b0   = blockIdx.x * 16; // first batch of this group

  // staging assignment: tid covers floats [tid*8, tid*8+8) of a 16x64 slot
  const int ms = tid >> 3;
  const int j0 = (tid & 7) * 8;

  // --- prologue: kick off async emission staging for t = 1..8
#pragma unroll
  for (int u = 1; u <= 8; ++u) {
    const float* g = &em[(size_t)(b0 + ms) * CRF_S * CRF_T + (size_t)u * CRF_T + j0];
    unsigned lds_off = (unsigned)(uintptr_t)&em_ring[u % 9][ms][j0];
    asm volatile("global_load_async_to_lds_b128 %0, %1, off\n\t"
                 "global_load_async_to_lds_b128 %0, %1, off offset:16"
                 :: "v"(lds_off), "v"(g) : "memory");
  }

  // --- B fragments: E[k][j] = exp(trans[k][j]), WMMA 16x16x32 f16 B layout
  v16h bE0, bE1;
#pragma unroll
  for (int e = 0; e < 16; ++e) {
    int k0 = (e < 8) ? (e + 8 * half) : (e + 8 + 8 * half);
    bE0[e] = (_Float16)__expf(trans[k0 * CRF_T + n]);
    bE1[e] = (_Float16)__expf(trans[(k0 + 32) * CRF_T + n]);
  }

  // --- D-layout register state: VGPR r holds row m = r + 8*half, column n.
  float a2[8];    // alpha / ln2
  float rmu[8];   // shift used for the p currently published
  const float stj = start_t[n];
#pragma unroll
  for (int r = 0; r < 8; ++r) {
    int m = r + 8 * half;
    a2[r] = (stj + em[(size_t)(b0 + m) * CRF_S * CRF_T + n]) * INV_LN2;  // t=0
  }

  // --- init publish (exact reduction once): pmax_sh[0], p_sh[0]
  {
    float pm[8];
#pragma unroll
    for (int r = 0; r < 8; ++r) pm[r] = a2[r];
#pragma unroll
    for (int d = 1; d < 16; d <<= 1) {
#pragma unroll
      for (int r = 0; r < 8; ++r) pm[r] = fmaxf(pm[r], __shfl_xor(pm[r], d, 32));
    }
    if (l15 == 0) {
#pragma unroll
      for (int r = 0; r < 8; ++r) pmax_sh[0][r + 8 * half][w] = pm[r];
    }
    __syncthreads();
#pragma unroll
    for (int r = 0; r < 8; ++r) {
      int m = r + 8 * half;
      float4 q = *(const float4*)&pmax_sh[0][m][0];
      rmu[r] = fmaxf(fmaxf(q.x, q.y), fmaxf(q.z, q.w));
      p_sh[0][m][n] = __float2half(fast_exp2(a2[r] - rmu[r]));
    }
    asm volatile("s_wait_asynccnt 0xe" ::: "memory");
    __syncthreads();
  }

  // mask byte for t=1, preloaded (all lanes load row l15 -> no divergence)
  bool predCur = mask[(size_t)(b0 + l15) * CRF_S + 1] != 0;

  for (int t = 1; t < CRF_S; ++t) {
    const int slot = t % 9;
    const int rb = (t - 1) & 1;  // read buffer
    const int wb = t & 1;        // write buffer

    // --- stale-by-one full row max (shift for this step's p publish)
    float rmn[8];
#pragma unroll
    for (int r = 0; r < 8; ++r) {
      float4 q = *(const float4*)&pmax_sh[rb][r + 8 * half][0];
      rmn[r] = fmaxf(fmaxf(q.x, q.y), fmaxf(q.z, q.w));
    }

    // --- emissions for this step from the LDS ring
    float emv[8];
#pragma unroll
    for (int r = 0; r < 8; ++r) emv[r] = em_ring[slot][r + 8 * half][n];

    unsigned mb = __builtin_amdgcn_ballot_w32(predCur);

    // --- A fragments from p_sh (16x32 f16 A layout; row = l15, K by half)
    v16h aLo, aHi;
    {
      const __half* row = &p_sh[rb][l15][0];
      ((float4*)&aLo)[0] = *(const float4*)(row + 8 * half);
      ((float4*)&aLo)[1] = *(const float4*)(row + 16 + 8 * half);
      ((float4*)&aHi)[0] = *(const float4*)(row + 32 + 8 * half);
      ((float4*)&aHi)[1] = *(const float4*)(row + 48 + 8 * half);
    }

    // two independent WMMAs (no D->C chain), combined with a dual-issue add
    v8f z = {};
    v8f accLo = __builtin_amdgcn_wmma_f32_16x16x32_f16(false, aLo, false, bE0,
                                                       (short)0, z, false, false);
    v8f accHi = __builtin_amdgcn_wmma_f32_16x16x32_f16(false, aHi, false, bE1,
                                                       (short)0, z, false, false);
    v8f acc = accLo + accHi;

    // --- refill ring: async stage emissions for t+8 (clamped tail)
    {
      int ts = (t + 8 < CRF_S) ? (t + 8) : (CRF_S - 1);
      const float* g =
          &em[(size_t)(b0 + ms) * CRF_S * CRF_T + (size_t)ts * CRF_T + j0];
      unsigned lds_off = (unsigned)(uintptr_t)&em_ring[(t + 8) % 9][ms][j0];
      asm volatile("global_load_async_to_lds_b128 %0, %1, off\n\t"
                   "global_load_async_to_lds_b128 %0, %1, off offset:16"
                   :: "v"(lds_off), "v"(g) : "memory");
    }
    // preload mask byte for t+1
    {
      int tm = (t + 1 < CRF_S) ? (t + 1) : t;
      predCur = mask[(size_t)(b0 + l15) * CRF_S + tm] != 0;
    }

    // --- base-2 alpha update: unconditional logs, branchless select
    float a2n[8];
#pragma unroll
    for (int r = 0; r < 8; ++r)
      a2n[r] = fmaf(emv[r], INV_LN2, rmu[r] + fast_log2(acc[r]));
#pragma unroll
    for (int r = 0; r < 8; ++r) {
      int m = r + 8 * half;
      a2[r] = ((mb >> m) & 1u) ? a2n[r] : a2[r];
    }

    // --- publish p immediately with the lagged shift (critical path)
#pragma unroll
    for (int r = 0; r < 8; ++r) {
      int m = r + 8 * half;
      p_sh[wb][m][n] = __float2half(fast_exp2(a2[r] - rmn[r]));
      rmu[r] = rmn[r];
    }

    // --- start next step's row-max reduction (off critical path)
    float pm[8];
#pragma unroll
    for (int r = 0; r < 8; ++r) pm[r] = a2[r];
#pragma unroll
    for (int d = 1; d < 16; d <<= 1) {
#pragma unroll
      for (int r = 0; r < 8; ++r) pm[r] = fmaxf(pm[r], __shfl_xor(pm[r], d, 32));
    }
    if (l15 == 0) {
#pragma unroll
      for (int r = 0; r < 8; ++r) pmax_sh[wb][r + 8 * half][w] = pm[r];
    }

    // next em slot resident before the single end-of-step barrier
    asm volatile("s_wait_asynccnt 0xe" ::: "memory");
    __syncthreads();
  }

  // --- final: partition[b] = logsumexp_j(a2[m][j]*ln2 + end[j])
  const float etj = end_t[n];
#pragma unroll
  for (int r = 0; r < 8; ++r) fin_sh[r + 8 * half][n] = a2[r] * LN2 + etj;
  __syncthreads();
  if (w == 0 && l < 16) {
    int m = l;
    float mx = -INFINITY;
    for (int j = 0; j < CRF_T; ++j) mx = fmaxf(mx, fin_sh[m][j]);
    float s = 0.f;
    for (int j = 0; j < CRF_T; ++j) s += __expf(fin_sh[m][j] - mx);
    partition[b0 + m] = mx + __logf(s);
  }
}

// ---------------------------------------------------------------------------
// Kernel 3: loss = mean(partition - score)
// ---------------------------------------------------------------------------
__global__ __launch_bounds__(128) void crf_loss_kernel(
    const float* __restrict__ score, const float* __restrict__ partition,
    float* __restrict__ out) {
  __shared__ float red[128];
  int tid = threadIdx.x;
  red[tid] = partition[tid] - score[tid];
  __syncthreads();
  for (int off = 64; off > 0; off >>= 1) {
    if (tid < off) red[tid] += red[tid + off];
    __syncthreads();
  }
  if (tid == 0) out[0] = red[0] / (float)CRF_B;
}

// ---------------------------------------------------------------------------
extern "C" void kernel_launch(void* const* d_in, const int* in_sizes, int n_in,
                              void* d_out, int out_size, void* d_ws, size_t ws_size,
                              hipStream_t stream) {
  (void)in_sizes; (void)n_in; (void)out_size; (void)ws_size;
  const float*         em      = (const float*)d_in[0];
  const int*           tags    = (const int*)d_in[1];
  const unsigned char* mask    = (const unsigned char*)d_in[2];
  const float*         trans   = (const float*)d_in[3];
  const float*         start_t = (const float*)d_in[4];
  const float*         end_t   = (const float*)d_in[5];

  float* score     = (float*)d_ws;   // [B]
  float* partition = score + CRF_B;  // [B]

  crf_score_kernel<<<CRF_B, 256, 0, stream>>>(em, tags, mask, trans, start_t,
                                              end_t, score);
  crf_partition_kernel<<<CRF_B / 16, 128, 0, stream>>>(em, mask, trans, start_t,
                                                       end_t, partition);
  crf_loss_kernel<<<1, 128, 0, stream>>>(score, partition, (float*)d_out);
}
